// FMA_15427522527280
// MI455X (gfx1250) — matrix operand order repaired and checked by reading, not compiled, not run on hardware
//
#include <hip/hip_runtime.h>
#include <math.h>

#define DMODEL 4096
#define SEQ    1024
#define BATCH  4

typedef __attribute__((ext_vector_type(16))) __bf16 v16bf;
typedef __attribute__((ext_vector_type(8)))  __bf16 v8bf;
typedef __attribute__((ext_vector_type(8)))  float  v8f;
typedef int v4i_vs __attribute__((__vector_size__(16)));
typedef __attribute__((address_space(1))) v4i_vs* gptr_v4i;
typedef __attribute__((address_space(3))) v4i_vs* lptr_v4i;

#if __has_builtin(__builtin_amdgcn_global_load_async_to_lds_b128)
#define HAVE_ASYNC_LDS 1
#else
#define HAVE_ASYNC_LDS 0
#endif

static __device__ __forceinline__ void wait_async_zero() {
#if HAVE_ASYNC_LDS
#if __has_builtin(__builtin_amdgcn_s_wait_asynccnt)
  __builtin_amdgcn_s_wait_asynccnt(0);
#else
  asm volatile("s_wait_asynccnt 0" ::: "memory");
#endif
#endif
}

static __device__ __forceinline__ unsigned short f2bf(float f) {
  unsigned u = __float_as_uint(f);
  unsigned r = u + 0x7FFFu + ((u >> 16) & 1u);   // round-to-nearest-even
  return (unsigned short)(r >> 16);
}
static __device__ __forceinline__ float bf2f(unsigned short h) {
  return __uint_as_float(((unsigned)h) << 16);
}

// ---------------------------------------------------------------- convert
__global__ void __launch_bounds__(256)
cvt_f32_bf16(const float* __restrict__ src, unsigned short* __restrict__ dst, long n) {
  long i = (long)blockIdx.x * blockDim.x + threadIdx.x;
  if (i < n) dst[i] = f2bf(src[i]);
}

// ---------------------------------------------------------------- WMMA GEMM
// Per batch z:  Y[m,n] = (sum_k A[m,k] * Bm[n,k]) * scale + bias[n] (+ bias2[z,n])
// A: M x Kd row-major bf16 (ldA), Bm: N x Kd row-major bf16 (ldB)  (Y = A * Bm^T)
// Block tile 128x128, K-step 32, double-buffered LDS staging via async Global->LDS.
// 8 waves: 4 along M x 2 along N; wave tile 32x64 (2x4 accumulators).

#define LDS_LD 40   // padded leading dim (bf16 elems) of a 128x32 LDS tile

// Stage one 128x32 bf16 tile (rows rowBase..rowBase+127, cols k0..k0+31) into LDS.
// 512 16-byte chunks; each of the 256 threads copies 2.
static __device__ __forceinline__ void stage_tile(const unsigned short* __restrict__ g,
                                                  int ld, int rowBase, int k0,
                                                  unsigned short* l, int tid) {
  #pragma unroll
  for (int h = 0; h < 2; h++) {
    const int c   = tid + h * 256;
    const int row = c >> 2;
    const int ko  = (c & 3) * 8;
    const unsigned short* gp = g + (long)(rowBase + row) * ld + (k0 + ko);
    unsigned short* lp = l + row * LDS_LD + ko;
#if HAVE_ASYNC_LDS
    __builtin_amdgcn_global_load_async_to_lds_b128(
        (gptr_v4i)(unsigned short*)gp, (lptr_v4i)lp, 0, 0);
#else
    v8bf v = *(const v8bf*)gp;
    *(v8bf*)lp = v;
#endif
  }
}

__global__ void __launch_bounds__(256)
wmma_gemm(const unsigned short* __restrict__ A, long aBatch, int ldA,
          const unsigned short* __restrict__ Bm, long bBatch, int ldB,
          int Kd,
          const float* __restrict__ bias,
          const float* __restrict__ bias2, long bias2Batch,
          float scale,
          float* __restrict__ outF, long ofBatch, int ldOF,
          unsigned short* __restrict__ outB, long obBatch, int ldOB,
          int transposeOut) {
  __shared__ unsigned short ldsA[2][128 * LDS_LD];
  __shared__ unsigned short ldsB[2][128 * LDS_LD];

  const int tid  = threadIdx.x;
  const int lane = tid & 31;
  const int wave = tid >> 5;
  const int wm = wave & 3;       // 4 waves along M (4*32 = 128)
  const int wn = wave >> 2;      // 2 waves along N (2*64 = 128)
  const int z  = blockIdx.z;
  const int mBlk = blockIdx.y * 128;
  const int nBlk = blockIdx.x * 128;

  const unsigned short* Ab = A  + (long)z * aBatch;
  const unsigned short* Bb = Bm + (long)z * bBatch;

  v8f acc[2][4] = {};

  const int lm   = lane & 15;            // row within 16-tile (A: M, B: N)
  const int lk8  = (lane >> 4) << 3;     // A half-wave K offset: 0 or 8
  const int lk16 = (lane >> 4) << 4;     // 0 or 16, unused (B uses lk8 like A)

  const int nsteps = Kd >> 5;

  stage_tile(Ab, ldA, mBlk, 0, ldsA[0], tid);
  stage_tile(Bb, ldB, nBlk, 0, ldsB[0], tid);
  wait_async_zero();
  __syncthreads();

  for (int s = 0; s < nsteps; s++) {
    const int cur = s & 1;
    if (s + 1 < nsteps) {
      const int k1 = (s + 1) << 5;
      stage_tile(Ab, ldA, mBlk, k1, ldsA[cur ^ 1], tid);
      stage_tile(Bb, ldB, nBlk, k1, ldsB[cur ^ 1], tid);
    }

    // Fragments from LDS (bank-conflict-avoiding padded rows).
    v16bf a[2], b[4];
    const unsigned short* la = ldsA[cur];
    const unsigned short* lb = ldsB[cur];
    #pragma unroll
    for (int i = 0; i < 2; i++) {
      const unsigned short* p = la + (wm * 32 + i * 16 + lm) * LDS_LD + lk8;
      v8bf lo = *(const v8bf*)p;
      v8bf hi = *(const v8bf*)(p + 16);
      #pragma unroll
      for (int e = 0; e < 8; e++) { a[i][e] = lo[e]; a[i][e + 8] = hi[e]; }
    }
    #pragma unroll
    for (int j = 0; j < 4; j++) {
      const unsigned short* p = lb + (wn * 64 + j * 16 + lm) * LDS_LD + lk8;
      v8bf lo = *(const v8bf*)p;
      v8bf hi = *(const v8bf*)(p + 16);
      #pragma unroll
      for (int e = 0; e < 8; e++) { b[j][e] = lo[e]; b[j][e + 8] = hi[e]; }
    }
    #pragma unroll
    for (int i = 0; i < 2; i++)
      #pragma unroll
      for (int j = 0; j < 4; j++)
        acc[i][j] = __builtin_amdgcn_wmma_f32_16x16x32_bf16(
            false, a[i], false, b[j], (short)0, acc[i][j], false, false);

    wait_async_zero();   // next buffer fully landed in LDS (this wave's share)
    __syncthreads();     // everyone done reading cur / staging next
  }

  // Epilogue.  C/D layout: n = lane%16; m = vgpr + 8*(lane/16).
  const int m0 = mBlk + wm * 32;
  const int n0 = nBlk + wn * 64;
  const int rbase = (lane >> 4) * 8;
  float*          oF = outF ? outF + (long)z * ofBatch : nullptr;
  unsigned short* oB = outB ? outB + (long)z * obBatch : nullptr;
  #pragma unroll
  for (int i = 0; i < 2; i++) {
    #pragma unroll
    for (int j = 0; j < 4; j++) {
      const int col = n0 + j * 16 + lm;
      float badd = 0.f;
      if (bias)  badd += bias[col];
      if (bias2) badd += bias2[(long)z * bias2Batch + col];
      #pragma unroll
      for (int e = 0; e < 8; e++) {
        const int row = m0 + i * 16 + rbase + e;
        const float v = acc[i][j][e] * scale + badd;
        if (oF) oF[(long)row * ldOF + col] = v;
        if (oB) {
          if (transposeOut) oB[(long)col * ldOB + row] = f2bf(v);
          else              oB[(long)row * ldOB + col] = f2bf(v);
        }
      }
    }
  }
}

// ---------------------------------------------------------------- softmax (rows of 1024)
__global__ void __launch_bounds__(256)
softmax_rows(const float* __restrict__ S, unsigned short* __restrict__ Sb, int n) {
  __shared__ float red[256];
  const long row = blockIdx.x;
  const float* r = S + row * (long)n;
  const int t = threadIdx.x;
  float vals[4];
  float mx = -3.4e38f;
  #pragma unroll
  for (int i = 0; i < 4; i++) { vals[i] = r[t + i * 256]; mx = fmaxf(mx, vals[i]); }
  red[t] = mx; __syncthreads();
  for (int s = 128; s > 0; s >>= 1) { if (t < s) red[t] = fmaxf(red[t], red[t + s]); __syncthreads(); }
  mx = red[0]; __syncthreads();
  float sum = 0.f;
  #pragma unroll
  for (int i = 0; i < 4; i++) { vals[i] = __expf(vals[i] - mx); sum += vals[i]; }
  red[t] = sum; __syncthreads();
  for (int s = 128; s > 0; s >>= 1) { if (t < s) red[t] += red[t + s]; __syncthreads(); }
  const float inv = 1.f / red[0];
  unsigned short* w = Sb + row * (long)n;
  #pragma unroll
  for (int i = 0; i < 4; i++) w[t + i * 256] = f2bf(vals[i] * inv);
}

// ---------------------------------------------------------------- depthwise conv reduction
// Vt: [b][c][t] bf16 (token-contiguous).  out[b*D + c] = sum_t Vt[b,c,t]*w[c,t] + bias[c]
__global__ void __launch_bounds__(256)
dwconv_reduce(const unsigned short* __restrict__ Vt, const float* __restrict__ w,
              const float* __restrict__ bias, float* __restrict__ out, int Dm, int Lk) {
  __shared__ float red[256];
  const int idx = blockIdx.x;          // b*Dm + c
  const int c = idx % Dm;
  const unsigned short* v = Vt + (long)idx * Lk;
  const float* wr = w + (long)c * Lk;
  float s = 0.f;
  for (int t = threadIdx.x; t < Lk; t += 256) s += bf2f(v[t]) * wr[t];
  red[threadIdx.x] = s; __syncthreads();
  for (int st = 128; st > 0; st >>= 1) { if (threadIdx.x < st) red[threadIdx.x] += red[threadIdx.x + st]; __syncthreads(); }
  if (threadIdx.x == 0) out[idx] = red[0] + bias[c];
}

// ---------------------------------------------------------------- launch
extern "C" void kernel_launch(void* const* d_in, const int* in_sizes, int n_in,
                              void* d_out, int out_size, void* d_ws, size_t ws_size,
                              hipStream_t stream) {
  (void)in_sizes; (void)n_in; (void)out_size; (void)ws_size;
  const float* x   = (const float*)d_in[0];
  const float* wq  = (const float*)d_in[1];
  const float* bq  = (const float*)d_in[2];
  const float* wk  = (const float*)d_in[3];
  const float* bk  = (const float*)d_in[4];
  const float* wv  = (const float*)d_in[5];
  const float* bv  = (const float*)d_in[6];
  // d_in[7]=ckw, d_in[8]=ckb unused: level-1 softmax is over a singleton axis => s1 == 1,
  // so out1 = depthwise_conv(V) only.
  const float* cvw = (const float*)d_in[9];
  const float* cvb = (const float*)d_in[10];
  float* out = (float*)d_out;

  const long nX = (long)BATCH * SEQ * DMODEL;   // 16,777,216
  const long nW = (long)DMODEL * DMODEL;        // 16,777,216

  char* ws = (char*)d_ws;
  unsigned short* xb  = (unsigned short*)ws; ws += nX * 2;
  unsigned short* wqb = (unsigned short*)ws; ws += nW * 2;
  unsigned short* wkb = (unsigned short*)ws; ws += nW * 2;
  unsigned short* wvb = (unsigned short*)ws; ws += nW * 2;
  unsigned short* Qb  = (unsigned short*)ws; ws += nX * 2;   // [b][t][c]
  unsigned short* Kb  = (unsigned short*)ws; ws += nX * 2;   // [b][t][c]
  unsigned short* Vtb = (unsigned short*)ws; ws += nX * 2;   // [b][c][t]  (transposed)
  float*          S   = (float*)ws;          ws += (long)BATCH * SEQ * SEQ * 4;
  unsigned short* Sb  = (unsigned short*)ws; ws += (long)BATCH * SEQ * SEQ * 2;
  float*          V1  = (float*)ws;          ws += (long)BATCH * DMODEL * 4;

  const int T = 256;
  cvt_f32_bf16<<<(unsigned)((nX + T - 1) / T), T, 0, stream>>>(x,  xb,  nX);
  cvt_f32_bf16<<<(unsigned)((nW + T - 1) / T), T, 0, stream>>>(wq, wqb, nW);
  cvt_f32_bf16<<<(unsigned)((nW + T - 1) / T), T, 0, stream>>>(wk, wkb, nW);
  cvt_f32_bf16<<<(unsigned)((nW + T - 1) / T), T, 0, stream>>>(wv, wvb, nW);

  // Projections: per batch M=SEQ, N=DMODEL, K=DMODEL;  Y = x * W^T + b
  dim3 gProj(DMODEL / 128, SEQ / 128, BATCH);
  wmma_gemm<<<gProj, T, 0, stream>>>(xb, (long)SEQ * DMODEL, DMODEL,
                                     wqb, 0L, DMODEL, DMODEL,
                                     bq, nullptr, 0L, 1.0f,
                                     nullptr, 0L, 0,
                                     Qb, (long)SEQ * DMODEL, DMODEL, 0);
  wmma_gemm<<<gProj, T, 0, stream>>>(xb, (long)SEQ * DMODEL, DMODEL,
                                     wkb, 0L, DMODEL, DMODEL,
                                     bk, nullptr, 0L, 1.0f,
                                     nullptr, 0L, 0,
                                     Kb, (long)SEQ * DMODEL, DMODEL, 0);
  wmma_gemm<<<gProj, T, 0, stream>>>(xb, (long)SEQ * DMODEL, DMODEL,
                                     wvb, 0L, DMODEL, DMODEL,
                                     bv, nullptr, 0L, 1.0f,
                                     nullptr, 0L, 0,
                                     Vtb, (long)DMODEL * SEQ, SEQ, 1);   // transposed V

  // out1 = depthwise conv of V (kernel == stride == SEQ)
  dwconv_reduce<<<BATCH * DMODEL, T, 0, stream>>>(Vtb, cvw, cvb, V1, DMODEL, SEQ);

  // Scores: S = (Q K^T) * 1/sqrt(D);  per batch M=N=SEQ, K=DMODEL
  dim3 gScore(SEQ / 128, SEQ / 128, BATCH);
  const float scale = 1.0f / sqrtf((float)DMODEL);
  wmma_gemm<<<gScore, T, 0, stream>>>(Qb, (long)SEQ * DMODEL, DMODEL,
                                      Kb, (long)SEQ * DMODEL, DMODEL, DMODEL,
                                      nullptr, nullptr, 0L, scale,
                                      S, (long)SEQ * SEQ, SEQ,
                                      nullptr, 0L, 0, 0);

  softmax_rows<<<BATCH * SEQ, T, 0, stream>>>(S, Sb, SEQ);

  // out = softmax(S) V + out1  (out1 broadcast folded into epilogue as bias2)
  dim3 gAttn(DMODEL / 128, SEQ / 128, BATCH);
  wmma_gemm<<<gAttn, T, 0, stream>>>(Sb, (long)SEQ * SEQ, SEQ,
                                     Vtb, (long)DMODEL * SEQ, SEQ, SEQ,
                                     nullptr, V1, (long)DMODEL, 1.0f,
                                     out, (long)SEQ * DMODEL, DMODEL,
                                     nullptr, 0L, 0, 0);
}
